// PinSAGELayer_23837068493398
// MI455X (gfx1250) — matrix operand order repaired and checked by reading, not compiled, not run on hardware
//
#include <hip/hip_runtime.h>
#include <hip/hip_bf16.h>

typedef __attribute__((ext_vector_type(16))) __bf16 v16bf;
typedef __attribute__((ext_vector_type(8)))  __bf16 v8bf;
typedef __attribute__((ext_vector_type(8)))  float  v8f;

#define DIN  128
#define DOUT 128

__device__ __forceinline__ unsigned short f2bf(float f) {
    unsigned int u = __float_as_uint(f);
    u += 0x7FFFu + ((u >> 16) & 1u);          // round-to-nearest-even
    return (unsigned short)(u >> 16);
}

// Build the 16-wide bf16 A fragment for v_wmma_f32_16x16x32_bf16.
// base points at (row n, k0 + 8*(lane>>4)); elements 0..7 come from +0..7,
// elements 8..15 from +16..23 (ISA 16-bit A 16x32 layout).
__device__ __forceinline__ v16bf load_a_frag(const unsigned short* base) {
    v8bf lo = *reinterpret_cast<const v8bf*>(base);
    v8bf hi = *reinterpret_cast<const v8bf*>(base + 16);
    v16bf a;
#pragma unroll
    for (int i = 0; i < 8; ++i) { a[i] = lo[i]; a[8 + i] = hi[i]; }
    return a;
}

// ---------------- conversion / init ----------------

__global__ void conv_bf16_kernel(const float* __restrict__ src,
                                 unsigned short* __restrict__ dst, int n) {
    int i = blockIdx.x * blockDim.x + threadIdx.x;
    if (i < n) dst[i] = f2bf(src[i]);
}

__global__ void init_kernel(float* __restrict__ agg, float* __restrict__ agg_w,
                            int n128, int n) {
    int i = blockIdx.x * blockDim.x + threadIdx.x;
    if (i < n128) agg[i] = 0.0f;
    if (i < n)    agg_w[i] = 1.0f;   // "+1" folded into init
}

// ---------------- GEMM1: z = relu(x @ Wl^T + bl), stored bf16 ----------------

__global__ void __launch_bounds__(256)
gemm1_kernel(const unsigned short* __restrict__ xb,
             const unsigned short* __restrict__ Wlb,
             const float* __restrict__ bl,
             unsigned short* __restrict__ zb) {
    const int lane = threadIdx.x & 31;
    const int wave = threadIdx.x >> 5;       // 8 waves -> 8 o-tiles
    const int n0   = blockIdx.x * 16;
    const int o0   = wave * 16;
    const int half = lane >> 4;
    const int lmod = lane & 15;
    const int n    = n0 + lmod;              // A row held by this lane
    const int o    = o0 + lmod;              // B column held by this lane

    const unsigned short* arow = xb  + (size_t)n * DIN + 8  * half;
    const unsigned short* brow = Wlb + (size_t)o * DIN + 8  * half;

    v8f c = {};
#pragma unroll
    for (int k0 = 0; k0 < DIN; k0 += 32) {
        v16bf a = load_a_frag(arow + k0);
        v16bf b = load_a_frag(brow + k0);
        c = __builtin_amdgcn_wmma_f32_16x16x32_bf16(false, a, false, b,
                                                    (short)0, c, false, false);
    }
    const float bias = bl[o];
#pragma unroll
    for (int r = 0; r < 8; ++r) {
        const int m = r + 8 * half;          // C/D layout: M = r + 8*(lane/16)
        float v = c[r] + bias;
        v = v > 0.0f ? v : 0.0f;
        zb[(size_t)(n0 + m) * DOUT + o] = f2bf(v);
    }
}

// ---------------- edge stage ----------------

__global__ void aggw_kernel(const long long* __restrict__ ei,
                            const float* __restrict__ w,
                            float* __restrict__ agg_w, int E) {
    int e = blockIdx.x * blockDim.x + threadIdx.x;
    if (e < E) atomicAdd(&agg_w[(int)ei[(size_t)E + e]], w[e]);
}

// one wave per edge: lane handles 4 contiguous channels (8B bf16 row read,
// 4 f32 atomic adds into the L2-resident agg buffer)
__global__ void __launch_bounds__(256)
edge_kernel(const long long* __restrict__ ei,
            const float* __restrict__ w,
            const unsigned short* __restrict__ zb,
            float* __restrict__ agg, int E) {
    const int wid = (blockIdx.x * blockDim.x + threadIdx.x) >> 5;
    if (wid >= E) return;
    const int lane = threadIdx.x & 31;
    const int s = (int)ei[wid];
    const int d = (int)ei[(size_t)E + wid];
    const float wt = w[wid];

    const unsigned int* zp =
        (const unsigned int*)(zb + (size_t)s * DOUT + lane * 4);
    unsigned int u0 = zp[0], u1 = zp[1];
    float f0 = __uint_as_float(u0 << 16);
    float f1 = __uint_as_float(u0 & 0xFFFF0000u);
    float f2 = __uint_as_float(u1 << 16);
    float f3 = __uint_as_float(u1 & 0xFFFF0000u);

    float* ap = agg + (size_t)d * DOUT + lane * 4;
    atomicAdd(ap + 0, wt * f0);
    atomicAdd(ap + 1, wt * f1);
    atomicAdd(ap + 2, wt * f2);
    atomicAdd(ap + 3, wt * f3);
}

__global__ void divconv_kernel(const float* __restrict__ agg,
                               const float* __restrict__ agg_w,
                               unsigned short* __restrict__ aggb, int total) {
    int i = blockIdx.x * blockDim.x + threadIdx.x;
    if (i < total) aggb[i] = f2bf(agg[i] / agg_w[i >> 7]);
}

// ---- GEMM2: out = normalize(relu(cat[x,agg] @ Wr^T + br)), fused epilogue ----

__global__ void __launch_bounds__(256)
gemm2_kernel(const unsigned short* __restrict__ xb,
             const unsigned short* __restrict__ aggb,
             const unsigned short* __restrict__ Wrb,
             const float* __restrict__ br,
             float* __restrict__ out) {
    __shared__ float rowsq[16];
    const int lane = threadIdx.x & 31;
    const int wave = threadIdx.x >> 5;
    const int n0   = blockIdx.x * 16;
    const int o0   = wave * 16;
    const int half = lane >> 4;
    const int lmod = lane & 15;
    const int n    = n0 + lmod;
    const int o    = o0 + lmod;

    const unsigned short* ax   = xb   + (size_t)n * DIN + 8 * half;
    const unsigned short* ag   = aggb + (size_t)n * DOUT + 8 * half;
    const unsigned short* brow = Wrb  + (size_t)o * (DIN + DOUT) + 8 * half;

    v8f c = {};
#pragma unroll
    for (int s = 0; s < 8; ++s) {
        const int k0 = s * 32;
        const unsigned short* abase = (s < 4) ? (ax + k0) : (ag + (k0 - DIN));
        v16bf a = load_a_frag(abase);
        v16bf b = load_a_frag(brow + k0);
        c = __builtin_amdgcn_wmma_f32_16x16x32_bf16(false, a, false, b,
                                                    (short)0, c, false, false);
    }

    if (threadIdx.x < 16) rowsq[threadIdx.x] = 0.0f;
    __syncthreads();

    const float bias = br[o];
    float vals[8];
#pragma unroll
    for (int r = 0; r < 8; ++r) {
        float v = c[r] + bias;
        v = v > 0.0f ? v : 0.0f;
        vals[r] = v;
        atomicAdd(&rowsq[r + 8 * half], v * v);   // LDS ds_add_f32
    }
    __syncthreads();

#pragma unroll
    for (int r = 0; r < 8; ++r) {
        const int m = r + 8 * half;
        const float inv = 1.0f / fmaxf(sqrtf(rowsq[m]), 1e-12f);
        out[(size_t)(n0 + m) * DOUT + o] = vals[r] * inv;
    }
}

// ---------------- launcher ----------------

extern "C" void kernel_launch(void* const* d_in, const int* in_sizes, int n_in,
                              void* d_out, int out_size, void* d_ws, size_t ws_size,
                              hipStream_t stream) {
    (void)n_in; (void)out_size; (void)ws_size;
    const float*     x  = (const float*)d_in[0];
    const long long* ei = (const long long*)d_in[1];
    const float*     w  = (const float*)d_in[2];
    const float*     Wl = (const float*)d_in[3];
    const float*     bl = (const float*)d_in[4];
    const float*     Wr = (const float*)d_in[5];
    const float*     br = (const float*)d_in[6];
    float*           out = (float*)d_out;

    const int N = in_sizes[0] / DIN;     // 100000
    const int E = in_sizes[2];           // 1600000

    // carve workspace (256B aligned): xb, zb, aggb (bf16), agg, agg_w (f32), Wlb, Wrb
    char* p = (char*)d_ws;
    auto carve = [&](size_t bytes) {
        char* r = p; p += (bytes + 255) & ~(size_t)255; return r;
    };
    unsigned short* xb    = (unsigned short*)carve((size_t)N * DIN * 2);
    unsigned short* zb    = (unsigned short*)carve((size_t)N * DOUT * 2);
    unsigned short* aggb  = (unsigned short*)carve((size_t)N * DOUT * 2);
    float*          agg   = (float*)carve((size_t)N * DOUT * 4);
    float*          agg_w = (float*)carve((size_t)N * 4);
    unsigned short* Wlb   = (unsigned short*)carve((size_t)DOUT * DIN * 2);
    unsigned short* Wrb   = (unsigned short*)carve((size_t)DOUT * (DIN + DOUT) * 2);

    const int T = 256;
    const int nTiles = N / 16;

    conv_bf16_kernel<<<(N * DIN + T - 1) / T, T, 0, stream>>>(x, xb, N * DIN);
    conv_bf16_kernel<<<(DOUT * DIN + T - 1) / T, T, 0, stream>>>(Wl, Wlb, DOUT * DIN);
    conv_bf16_kernel<<<(DOUT * (DIN + DOUT) + T - 1) / T, T, 0, stream>>>(
        Wr, Wrb, DOUT * (DIN + DOUT));
    init_kernel<<<(N * DOUT + T - 1) / T, T, 0, stream>>>(agg, agg_w, N * DOUT, N);

    gemm1_kernel<<<nTiles, 256, 0, stream>>>(xb, Wlb, bl, zb);

    aggw_kernel<<<(E + T - 1) / T, T, 0, stream>>>(ei, w, agg_w, E);
    edge_kernel<<<(E * 32 + T - 1) / T, T, 0, stream>>>(ei, w, zb, agg, E);
    divconv_kernel<<<(N * DOUT + T - 1) / T, T, 0, stream>>>(agg, agg_w, aggb, N * DOUT);

    gemm2_kernel<<<nTiles, 256, 0, stream>>>(xb, aggb, Wrb, br, out);
}
